// GroupedQueryAttention_78666620993897
// MI455X (gfx1250) — matrix operand order repaired and checked by reading, not compiled, not run on hardware
//
#include <hip/hip_runtime.h>

// ---------------------------------------------------------------------------
// GQA for MI455X (gfx1250): all matmuls on V_WMMA_F32_16X16X32_BF16.
// Pipeline: cvt(f32->bf16) -> QKV proj GEMMs -> LN+RoPE -> V transpose ->
//           flash attention (WMMA QK^T + online softmax + WMMA PV) -> O GEMM.
// GEMM v2: 32x32 register tile per wave (2x2 WMMA) for 2x fragment reuse.
// ---------------------------------------------------------------------------

#define B_    4
#define S_    1024
#define DIM_  1024
#define MS_   (B_ * S_)        // 4096 rows
#define KVD_  256              // GROUPS*HD
#define HD_   64
#define PW_   32
#define SCALE_ 0.125f          // 1/sqrt(64)

typedef __attribute__((ext_vector_type(8)))  __bf16 v8bf;
typedef __attribute__((ext_vector_type(16))) __bf16 v16bf;
typedef __attribute__((ext_vector_type(8)))  float  v8f;

#define WMMA_BF16(a, b, c) \
  __builtin_amdgcn_wmma_f32_16x16x32_bf16(false, (a), false, (b), (short)0, (c), false, false)

__device__ __forceinline__ __bf16 f2bf(float f) {
  unsigned int u = __builtin_bit_cast(unsigned int, f);
  u += 0x7fffu + ((u >> 16) & 1u);          // round-to-nearest-even
  unsigned short s = (unsigned short)(u >> 16);
  return __builtin_bit_cast(__bf16, s);
}

__device__ __forceinline__ v8f zero8() {
  v8f z;
#pragma unroll
  for (int i = 0; i < 8; ++i) z[i] = 0.0f;
  return z;
}

// A-fragment (16x32 bf16, MxK): lane L<16 -> row L, K = {0..7, 16..23};
// lanes 16..31 -> same rows, K shifted by +8.  rowk0 points at (row, k0).
__device__ __forceinline__ v16bf load_frag_a(const __bf16* rowk0, int lane) {
  const __bf16* p = rowk0 + ((lane >> 4) << 3);
  v8bf c0 = *(const v8bf*)(p);
  v8bf c1 = *(const v8bf*)(p + 16);
  v16bf a;
#pragma unroll
  for (int i = 0; i < 8; ++i) { a[i] = c0[i]; a[i + 8] = c1[i]; }
  return a;
}

// B-fragment (32x16 bf16, KxN): lane L<16 -> col L, K = {0..7, 16..23}; lanes 16..31 ->
// same cols, K shifted by +8.  colk0 points at column base (contiguous in K) at k0.
__device__ __forceinline__ v16bf load_frag_b(const __bf16* colk0, int lane) {
  const __bf16* p = colk0 + ((lane >> 4) << 3);
  v8bf c0 = *(const v8bf*)(p);
  v8bf c1 = *(const v8bf*)(p + 16);
  v16bf b;
#pragma unroll
  for (int i = 0; i < 8; ++i) { b[i] = c0[i]; b[i + 8] = c1[i]; }
  return b;
}

// ---------------------------------------------------------------------------
// f32 -> bf16 elementwise convert
// ---------------------------------------------------------------------------
__global__ void cvt_bf16_kernel(const float* __restrict__ src,
                                __bf16* __restrict__ dst, int n) {
  int i = blockIdx.x * blockDim.x + threadIdx.x;
  if (i < n) dst[i] = f2bf(src[i]);
}

// ---------------------------------------------------------------------------
// C[M,N] = A[M,K] @ W[N,K]^T + bias[N]   (A,W bf16 row-major; C f32)
// block = 256 threads = 8 waves arranged 4(M) x 2(N); wave owns a 32x32 tile
// (2x2 WMMA accumulators) -> each fragment load feeds two WMMAs.
// Block tile: 128(M) x 64(N).
// ---------------------------------------------------------------------------
__global__ __launch_bounds__(256) void gemm_bf16_nt_kernel(
    const __bf16* __restrict__ A, const __bf16* __restrict__ W,
    const float* __restrict__ bias, float* __restrict__ C,
    int M, int N, int K) {
  const int lane = threadIdx.x & 31;
  const int wave = threadIdx.x >> 5;
  const int wm = wave >> 1;                 // 0..3
  const int wn = wave & 1;                  // 0..1
  const int m0 = blockIdx.y * 128 + wm * 32;
  const int n0 = blockIdx.x * 64 + wn * 32;
  if (m0 >= M || n0 >= N) return;           // wave-uniform: EXEC stays all-ones

  const int l15 = lane & 15;
  const __bf16* arow0 = A + (size_t)(m0 + l15) * K;
  const __bf16* arow1 = A + (size_t)(m0 + 16 + l15) * K;
  const __bf16* wrow0 = W + (size_t)(n0 + l15) * K;
  const __bf16* wrow1 = W + (size_t)(n0 + 16 + l15) * K;

  v8f acc00 = zero8(), acc01 = zero8(), acc10 = zero8(), acc11 = zero8();
  for (int k0 = 0; k0 < K; k0 += 32) {
    __builtin_prefetch(arow0 + k0 + 512, 0, 0);   // global_prefetch_b8
    __builtin_prefetch(wrow0 + k0 + 512, 0, 0);
    v16bf a0 = load_frag_a(arow0 + k0, lane);
    v16bf a1 = load_frag_a(arow1 + k0, lane);
    v16bf b0 = load_frag_b(wrow0 + k0, lane);
    v16bf b1 = load_frag_b(wrow1 + k0, lane);
    acc00 = WMMA_BF16(a0, b0, acc00);
    acc01 = WMMA_BF16(a0, b1, acc01);
    acc10 = WMMA_BF16(a1, b0, acc10);
    acc11 = WMMA_BF16(a1, b1, acc11);
  }

  const float bc0 = bias[n0 + l15];
  const float bc1 = bias[n0 + 16 + l15];
  const int ro = (lane >> 4) * 8;
#pragma unroll
  for (int r = 0; r < 8; ++r) {
    C[(size_t)(m0 + ro + r) * N + (n0 + l15)]           = acc00[r] + bc0;
    C[(size_t)(m0 + ro + r) * N + (n0 + 16 + l15)]      = acc01[r] + bc1;
    C[(size_t)(m0 + 16 + ro + r) * N + (n0 + l15)]      = acc10[r] + bc0;
    C[(size_t)(m0 + 16 + ro + r) * N + (n0 + 16 + l15)] = acc11[r] + bc1;
  }
}

// ---------------------------------------------------------------------------
// Per-head LayerNorm(64) + rotary on first 32 dims, f32 in -> bf16 out.
// One wave per (token, head); lane owns dims {2*lane, 2*lane+1}.
// ---------------------------------------------------------------------------
__global__ __launch_bounds__(256) void ln_rope_kernel(
    const float* __restrict__ src, __bf16* __restrict__ dst,
    const int* __restrict__ time_id, const float* __restrict__ gamma,
    const float* __restrict__ beta, int heads, int rowlen, int ntok) {
  const int lane = threadIdx.x & 31;
  const int wave = threadIdx.x >> 5;
  const int task = blockIdx.x * 8 + wave;
  if (task >= ntok * heads) return;
  const int tok = task / heads;
  const int h   = task % heads;

  const float* p = src + (size_t)tok * rowlen + h * 64;
  const int d0 = lane * 2;
  float x0 = p[d0], x1 = p[d0 + 1];

  float s = x0 + x1;
#pragma unroll
  for (int off = 16; off >= 1; off >>= 1) s += __shfl_xor(s, off, 32);
  const float mean = s * (1.0f / 64.0f);
  const float dx0 = x0 - mean, dx1 = x1 - mean;
  float v = dx0 * dx0 + dx1 * dx1;
#pragma unroll
  for (int off = 16; off >= 1; off >>= 1) v += __shfl_xor(v, off, 32);
  const float rstd = rsqrtf(v * (1.0f / 64.0f) + 1e-5f);

  float y0 = dx0 * rstd * gamma[d0] + beta[d0];
  float y1 = dx1 * rstd * gamma[d0 + 1] + beta[d0 + 1];

  if (d0 < PW_) {                                   // rotary on first 32 dims
    const float t = (float)time_id[tok];
    // theta = BASE^(-d0/PW) = exp(-d0/32 * ln(10000))
    const float theta = __expf((float)d0 * (-9.210340371976184f / 32.0f));
    const float a = t * theta;
    const float c = __cosf(a), sn = __sinf(a);
    const float r0 = c * y0 - sn * y1;
    const float r1 = c * y1 + sn * y0;
    y0 = r0; y1 = r1;
  }
  __bf16* q = dst + (size_t)tok * rowlen + h * 64 + d0;
  q[0] = f2bf(y0);
  q[1] = f2bf(y1);
}

// ---------------------------------------------------------------------------
// v f32 [B,S,256] -> vt bf16 [B][4][64][S]  (so PV B-fragments are contiguous)
// ---------------------------------------------------------------------------
__global__ void transpose_v_kernel(const float* __restrict__ v,
                                   __bf16* __restrict__ vt) {
  const int i = blockIdx.x * blockDim.x + threadIdx.x;   // B*4*64*S
  const int s = i & (S_ - 1);
  const int d = (i >> 10) & 63;
  const int g = (i >> 16) & 3;
  const int b = i >> 18;
  vt[i] = f2bf(v[((size_t)(b * S_ + s)) * KVD_ + g * HD_ + d]);
}

// ---------------------------------------------------------------------------
// Flash attention: one wave per (b, g, h, 16-query tile).
// QK^T via WMMA, online softmax (shfl_xor row reductions), P relayout via
// LDS (C-layout -> A-frag), PV via WMMA into f32 accumulators.
// ---------------------------------------------------------------------------
__global__ __launch_bounds__(256) void gqa_attn_kernel(
    const __bf16* __restrict__ qb, const __bf16* __restrict__ kb,
    const __bf16* __restrict__ vt, const float* __restrict__ var_emb,
    const int* __restrict__ qvar, const int* __restrict__ kvar,
    __bf16* __restrict__ ob) {
  __shared__ __align__(16) __bf16 plds[8][16 * 32];
  const int lane = threadIdx.x & 31;
  const int wave = threadIdx.x >> 5;
  const int task = blockIdx.x * 8 + wave;    // 4096 tasks exactly
  const int qt = task & 63;
  const int h  = (task >> 6) & 3;
  const int g  = (task >> 8) & 3;
  const int b  = task >> 10;
  const int hg = g * 4 + h;
  const int q0 = qt * 16;
  const int hi  = lane >> 4;
  const int l15 = lane & 15;

  // Q A-fragments for both 32-wide head-dim chunks (reused for all KV steps)
  const __bf16* qrow = qb + ((size_t)(b * S_ + q0 + l15) * DIM_) + hg * HD_;
  const v16bf aq0 = load_frag_a(qrow, lane);
  const v16bf aq1 = load_frag_a(qrow + 32, lane);

  const float w0 = var_emb[hg];        // var_emb[0, g*4+h]
  const float w1 = var_emb[16 + hg];   // var_emb[1, g*4+h]

  float mrow[8], lrow[8];
  int   qv[8];
  v8f o0 = zero8(), o1 = zero8(), o2 = zero8(), o3 = zero8();
#pragma unroll
  for (int r = 0; r < 8; ++r) {
    mrow[r] = -1e30f;
    lrow[r] = 0.0f;
    qv[r] = qvar[b * S_ + q0 + hi * 8 + r];
  }

  for (int kv0 = 0; kv0 < S_; kv0 += 32) {
    // ---- scores: two 16x16 tiles covering kv columns [kv0, kv0+32) ----
    v8f c0 = zero8(), c1 = zero8();
    {
      const __bf16* kcol = kb + ((size_t)(b * S_ + kv0 + l15) * KVD_) + g * HD_;
      c0 = WMMA_BF16(aq0, load_frag_b(kcol, lane), c0);
      c0 = WMMA_BF16(aq1, load_frag_b(kcol + 32, lane), c0);
    }
    {
      const __bf16* kcol = kb + ((size_t)(b * S_ + kv0 + 16 + l15) * KVD_) + g * HD_;
      c1 = WMMA_BF16(aq0, load_frag_b(kcol, lane), c1);
      c1 = WMMA_BF16(aq1, load_frag_b(kcol + 32, lane), c1);
    }

    // ---- scale + variable-id bias, row max ----
    const int kvv0 = kvar[b * S_ + kv0 + l15];
    const int kvv1 = kvar[b * S_ + kv0 + 16 + l15];
    float s0[8], s1[8], rmax[8];
#pragma unroll
    for (int r = 0; r < 8; ++r) {
      s0[r] = c0[r] * SCALE_ + ((qv[r] == kvv0) ? w1 : w0);
      s1[r] = c1[r] * SCALE_ + ((qv[r] == kvv1) ? w1 : w0);
      float rm = fmaxf(s0[r], s1[r]);
#pragma unroll
      for (int off = 8; off >= 1; off >>= 1)
        rm = fmaxf(rm, __shfl_xor(rm, off, 16));
      rmax[r] = rm;
    }

    // ---- online softmax update; write P (bf16) to LDS in row-major ----
#pragma unroll
    for (int r = 0; r < 8; ++r) {
      const float nm = fmaxf(mrow[r], rmax[r]);
      const float sc = __expf(mrow[r] - nm);
      mrow[r] = nm;
      const float p0 = __expf(s0[r] - nm);
      const float p1 = __expf(s1[r] - nm);
      plds[wave][(hi * 8 + r) * 32 + l15]      = f2bf(p0);
      plds[wave][(hi * 8 + r) * 32 + 16 + l15] = f2bf(p1);
      float rs = p0 + p1;
#pragma unroll
      for (int off = 8; off >= 1; off >>= 1)
        rs += __shfl_xor(rs, off, 16);
      lrow[r] = lrow[r] * sc + rs;
      o0[r] *= sc; o1[r] *= sc; o2[r] *= sc; o3[r] *= sc;
    }

    asm volatile("s_wait_dscnt 0" ::: "memory");  // P stores -> A-frag reads

    // ---- PV: P(16x32) @ V(32x64) via 4 WMMAs (V pre-transposed) ----
    const v16bf pa = load_frag_a(&plds[wave][l15 * 32], lane);
    const __bf16* vbase = vt + (((size_t)(b * 4 + g) * HD_) * S_) + kv0;
    o0 = WMMA_BF16(pa, load_frag_b(vbase + (size_t)(0  + l15) * S_, lane), o0);
    o1 = WMMA_BF16(pa, load_frag_b(vbase + (size_t)(16 + l15) * S_, lane), o1);
    o2 = WMMA_BF16(pa, load_frag_b(vbase + (size_t)(32 + l15) * S_, lane), o2);
    o3 = WMMA_BF16(pa, load_frag_b(vbase + (size_t)(48 + l15) * S_, lane), o3);
  }

  // ---- normalize and write bf16 output [B,S,DIM], head at hg*64 ----
  __bf16* obase = ob + ((size_t)(b * S_)) * DIM_ + hg * HD_;
#pragma unroll
  for (int r = 0; r < 8; ++r) {
    const float inv = 1.0f / lrow[r];
    __bf16* op = obase + (size_t)(q0 + hi * 8 + r) * DIM_ + l15;
    op[0]  = f2bf(o0[r] * inv);
    op[16] = f2bf(o1[r] * inv);
    op[32] = f2bf(o2[r] * inv);
    op[48] = f2bf(o3[r] * inv);
  }
}

// ---------------------------------------------------------------------------
// Host launcher
// ---------------------------------------------------------------------------
extern "C" void kernel_launch(void* const* d_in, const int* in_sizes, int n_in,
                              void* d_out, int out_size, void* d_ws, size_t ws_size,
                              hipStream_t stream) {
  const float* x     = (const float*)d_in[0];
  const float* kv    = (const float*)d_in[1];
  const float* q_w   = (const float*)d_in[2];
  const float* q_b   = (const float*)d_in[3];
  const float* k_w   = (const float*)d_in[4];
  const float* k_b   = (const float*)d_in[5];
  const float* v_w   = (const float*)d_in[6];
  const float* v_b   = (const float*)d_in[7];
  const float* o_w   = (const float*)d_in[8];
  const float* o_b   = (const float*)d_in[9];
  const float* qn_g  = (const float*)d_in[10];
  const float* qn_b  = (const float*)d_in[11];
  const float* kn_g  = (const float*)d_in[12];
  const float* kn_b  = (const float*)d_in[13];
  const float* v_emb = (const float*)d_in[14];
  const int*   qvar  = (const int*)d_in[15];
  const int*   kvar  = (const int*)d_in[16];
  const int*   qtime = (const int*)d_in[17];
  const int*   ktime = (const int*)d_in[18];
  float* out = (float*)d_out;

  char* ws = (char*)d_ws;
  size_t off = 0;
  auto take = [&](size_t bytes) -> char* {
    char* p = ws + off;
    off += (bytes + 255) & ~(size_t)255;
    return p;
  };

  __bf16* x_bf   = (__bf16*)take((size_t)MS_ * DIM_ * 2);
  __bf16* kv_bf  = (__bf16*)take((size_t)MS_ * DIM_ * 2);
  __bf16* qw_bf  = (__bf16*)take((size_t)DIM_ * DIM_ * 2);
  __bf16* kw_bf  = (__bf16*)take((size_t)KVD_ * DIM_ * 2);
  __bf16* vw_bf  = (__bf16*)take((size_t)KVD_ * DIM_ * 2);
  __bf16* ow_bf  = (__bf16*)take((size_t)DIM_ * DIM_ * 2);
  float*  q_f    = (float*)take((size_t)MS_ * DIM_ * 4);
  float*  k_f    = (float*)take((size_t)MS_ * KVD_ * 4);
  float*  v_f    = (float*)take((size_t)MS_ * KVD_ * 4);
  __bf16* q_bf   = (__bf16*)take((size_t)MS_ * DIM_ * 2);
  __bf16* k_bf   = (__bf16*)take((size_t)MS_ * KVD_ * 2);
  __bf16* vt_bf  = (__bf16*)take((size_t)MS_ * KVD_ * 2);
  __bf16* at_bf  = (__bf16*)take((size_t)MS_ * DIM_ * 2);
  if (off > ws_size) return;   // workspace too small: bail deterministically

  const int TPB = 256;
  // 1) bf16 conversions
  cvt_bf16_kernel<<<(MS_ * DIM_ + TPB - 1) / TPB, TPB, 0, stream>>>(x, x_bf, MS_ * DIM_);
  cvt_bf16_kernel<<<(MS_ * DIM_ + TPB - 1) / TPB, TPB, 0, stream>>>(kv, kv_bf, MS_ * DIM_);
  cvt_bf16_kernel<<<(DIM_ * DIM_ + TPB - 1) / TPB, TPB, 0, stream>>>(q_w, qw_bf, DIM_ * DIM_);
  cvt_bf16_kernel<<<(KVD_ * DIM_ + TPB - 1) / TPB, TPB, 0, stream>>>(k_w, kw_bf, KVD_ * DIM_);
  cvt_bf16_kernel<<<(KVD_ * DIM_ + TPB - 1) / TPB, TPB, 0, stream>>>(v_w, vw_bf, KVD_ * DIM_);
  cvt_bf16_kernel<<<(DIM_ * DIM_ + TPB - 1) / TPB, TPB, 0, stream>>>(o_w, ow_bf, DIM_ * DIM_);

  // 2) Q/K/V projection GEMMs (WMMA, 32x32 per wave, block tile 128x64)
  gemm_bf16_nt_kernel<<<dim3(DIM_ / 64, MS_ / 128), TPB, 0, stream>>>(
      x_bf, qw_bf, q_b, q_f, MS_, DIM_, DIM_);
  gemm_bf16_nt_kernel<<<dim3(KVD_ / 64, MS_ / 128), TPB, 0, stream>>>(
      kv_bf, kw_bf, k_b, k_f, MS_, KVD_, DIM_);
  gemm_bf16_nt_kernel<<<dim3(KVD_ / 64, MS_ / 128), TPB, 0, stream>>>(
      kv_bf, vw_bf, v_b, v_f, MS_, KVD_, DIM_);

  // 3) LayerNorm + rotary -> bf16
  ln_rope_kernel<<<(MS_ * 16) / 8, TPB, 0, stream>>>(
      q_f, q_bf, qtime, qn_g, qn_b, 16, DIM_, MS_);
  ln_rope_kernel<<<(MS_ * 4) / 8, TPB, 0, stream>>>(
      k_f, k_bf, ktime, kn_g, kn_b, 4, KVD_, MS_);

  // 4) V transpose -> [b][g][d][s] bf16
  transpose_v_kernel<<<(MS_ * KVD_) / TPB, TPB, 0, stream>>>(v_f, vt_bf);

  // 5) flash attention (4096 wave-tasks, 8 waves/block)
  gqa_attn_kernel<<<(B_ * 4 * 4 * (S_ / 16)) / 8, TPB, 0, stream>>>(
      q_bf, k_bf, vt_bf, v_emb, qvar, kvar, at_bf);

  // 6) output projection GEMM -> f32 d_out
  gemm_bf16_nt_kernel<<<dim3(DIM_ / 64, MS_ / 128), TPB, 0, stream>>>(
      at_bf, ow_bf, o_b, out, MS_, DIM_, DIM_);
}
